// OptimizedMoE_83133386982056
// MI455X (gfx1250) — compile-verified, matrix operand order checked by reading, not run on hardware
//
#include <hip/hip_runtime.h>

#define BB 4
#define SS 2048
#define DD 2048
#define EE 8
#define BS (BB * SS)          // 8192 tokens

#define TM 128
#define TN 128
#define TK 32
#define LDT 40                // padded LDS row stride in bf16 elements (80B rows, 16B-aligned segments)
#define NK (DD / TK)          // 64 K-steps

typedef __attribute__((ext_vector_type(4)))  __bf16 v4bf;
typedef __attribute__((ext_vector_type(8)))  __bf16 v8bf;
typedef __attribute__((ext_vector_type(16))) __bf16 v16bf;
typedef __attribute__((ext_vector_type(8)))  float  v8f;

#if defined(__gfx1250__) && __has_builtin(__builtin_amdgcn_global_load_async_to_lds_b128) && \
    __has_builtin(__builtin_amdgcn_s_wait_asynccnt)
#define HAS_ASYNC 1
typedef __attribute__((__vector_size__(16))) int vsi4;
typedef __attribute__((address_space(1))) vsi4 gvsi4;   // global int4 (builtin's gaddr param type)
typedef __attribute__((address_space(3))) vsi4 lvsi4;   // LDS int4 (builtin's lds param type)
#else
#define HAS_ASYNC 0
#endif

static __device__ __forceinline__ v16bf cat8(v8bf a, v8bf b) {
    return __builtin_shufflevector(a, b, 0,1,2,3,4,5,6,7,8,9,10,11,12,13,14,15);
}

// ---------------------------------------------------------------------------
// Kernel 1: convert x (f32) -> bf16, 4 elements per thread
// ---------------------------------------------------------------------------
__global__ void moe_conv_x(const float* __restrict__ x, __bf16* __restrict__ xb) {
    size_t i = ((size_t)blockIdx.x * blockDim.x + threadIdx.x) * 4;
    float4 v = *(const float4*)(x + i);
    v4bf o = { (__bf16)v.x, (__bf16)v.y, (__bf16)v.z, (__bf16)v.w };
    *(v4bf*)(xb + i) = o;
}

// ---------------------------------------------------------------------------
// Kernel 2: expert_w [E][D][F] f32 -> wT [E][F][D] bf16 (tiled LDS transpose)
// ---------------------------------------------------------------------------
__global__ void moe_conv_wT(const float* __restrict__ w, __bf16* __restrict__ wT) {
    __shared__ __bf16 tile[32][33];
    int e  = blockIdx.z;
    int f0 = blockIdx.x * 32;
    int d0 = blockIdx.y * 32;
    int tx = threadIdx.x & 31;
    int ty = threadIdx.x >> 5;     // 0..7
    const float* we  = w  + (size_t)e * DD * DD;
    __bf16*      wTe = wT + (size_t)e * DD * DD;
#pragma unroll
    for (int i = 0; i < 4; ++i) {
        int d = d0 + ty + i * 8;
        tile[ty + i * 8][tx] = (__bf16)we[(size_t)d * DD + f0 + tx];
    }
    __syncthreads();
#pragma unroll
    for (int i = 0; i < 4; ++i) {
        int f = f0 + ty + i * 8;
        wTe[(size_t)f * DD + d0 + tx] = tile[tx][ty + i * 8];
    }
}

// ---------------------------------------------------------------------------
// Kernel 3: router — one wave (32 lanes) per token.
// ---------------------------------------------------------------------------
__global__ void moe_router(const float* __restrict__ x,
                           const float* __restrict__ rw,
                           const float* __restrict__ rb,
                           int* __restrict__ counts,
                           int* __restrict__ tok_list,
                           float* __restrict__ gate_list) {
    int lane = threadIdx.x & 31;
    int wave = threadIdx.x >> 5;
    int tok  = blockIdx.x * 8 + wave;

    float acc[EE];
#pragma unroll
    for (int e = 0; e < EE; ++e) acc[e] = 0.f;

    const float* xr = x + (size_t)tok * DD;
    for (int d = lane; d < DD; d += 32) {
        float xv = xr[d];
        const float4* rv = (const float4*)(rw + (size_t)d * EE);
        float4 a = rv[0], b = rv[1];
        acc[0] += xv * a.x; acc[1] += xv * a.y; acc[2] += xv * a.z; acc[3] += xv * a.w;
        acc[4] += xv * b.x; acc[5] += xv * b.y; acc[6] += xv * b.z; acc[7] += xv * b.w;
    }
#pragma unroll
    for (int off = 16; off > 0; off >>= 1) {
#pragma unroll
        for (int e = 0; e < EE; ++e)
            acc[e] += __shfl_xor(acc[e], off, 32);
    }

    if (lane == 0) {
#pragma unroll
        for (int e = 0; e < EE; ++e) acc[e] += rb[e];
        int i0 = 0;
#pragma unroll
        for (int e = 1; e < EE; ++e) if (acc[e] > acc[i0]) i0 = e;
        int i1 = (i0 == 0) ? 1 : 0;
#pragma unroll
        for (int e = 0; e < EE; ++e) if (e != i0 && acc[e] > acc[i1]) i1 = e;
        float w0 = 1.f / (1.f + __expf(acc[i1] - acc[i0]));
        float w1 = 1.f - w0;
        int p0 = atomicAdd(&counts[i0], 1);
        tok_list[i0 * BS + p0]  = tok;
        gate_list[i0 * BS + p0] = w0;
        int p1 = atomicAdd(&counts[i1], 1);
        tok_list[i1 * BS + p1]  = tok;
        gate_list[i1 * BS + p1] = w1;
    }
}

// ---------------------------------------------------------------------------
// Kernel 4: grouped expert GEMM, WMMA bf16 + async global->LDS double buffer.
// Block tile 128x128, K-step 32. 8 waves = 2(m) x 4(n); each wave 64x32 =
// 4x2 fragments -> 8 v_wmma_f32_16x16x32_bf16 per K-step per wave.
// Fill of K-step k+1 (ASYNCcnt-tracked) overlaps WMMA of K-step k.
// ---------------------------------------------------------------------------
__global__ __launch_bounds__(256)
void moe_gemm(const __bf16* __restrict__ xb, const __bf16* __restrict__ wT,
              const int* __restrict__ counts, const int* __restrict__ tok_list,
              const float* __restrict__ gate_list, const float* __restrict__ eb,
              float* __restrict__ out) {
    int e   = blockIdx.z;
    int cnt = counts[e];
    int m0  = blockIdx.y * TM;
    if (m0 >= cnt) return;
    int n0  = blockIdx.x * TN;

    __shared__ __bf16 sA[2][TM * LDT];
    __shared__ __bf16 sB[2][TN * LDT];
    __shared__ int    stok[TM];
    __shared__ float  sgate[TM];

    int tid  = threadIdx.x;
    int lane = tid & 31;
    int wave = tid >> 5;
    int wm   = wave & 1;        // 0..1 -> 64-row slice
    int wn   = wave >> 1;       // 0..3 -> 32-col slice
    int half = lane >> 4;       // K-chunk select (WMMA 16-bit operand layout)
    int r16  = lane & 15;

    if (tid < TM) {
        int rg = m0 + tid;
        if (rg < cnt) {
            stok[tid]  = tok_list[e * BS + rg];
            sgate[tid] = gate_list[e * BS + rg];
        } else {
            stok[tid]  = 0;
            sgate[tid] = 0.f;
        }
    }
    __syncthreads();

    // Loop-invariant gather addresses: this thread fills two 16B segments of
    // A and two of B per K-step.  flat = tid*2 + i, row = flat/4, seg = flat%4.
    const __bf16* wTe = wT + (size_t)e * DD * DD;
    int flat0 = tid * 2,     row0 = flat0 >> 2, seg0 = flat0 & 3;
    int flat1 = tid * 2 + 1, row1 = flat1 >> 2, seg1 = flat1 & 3;
    const __bf16* gA0 = xb  + (size_t)stok[row0] * DD + seg0 * 8;
    const __bf16* gA1 = xb  + (size_t)stok[row1] * DD + seg1 * 8;
    const __bf16* gB0 = wTe + (size_t)(n0 + row0) * DD + seg0 * 8;
    const __bf16* gB1 = wTe + (size_t)(n0 + row1) * DD + seg1 * 8;
    int offA0 = row0 * LDT + seg0 * 8;
    int offA1 = row1 * LDT + seg1 * 8;

    v8f acc[4][2];
#pragma unroll
    for (int m = 0; m < 4; ++m)
#pragma unroll
        for (int n = 0; n < 2; ++n)
            acc[m][n] = (v8f){0.f,0.f,0.f,0.f,0.f,0.f,0.f,0.f};

    // ---- fill helper (async when available, sync fallback otherwise) ----
    auto fill = [&](int buf, int k) {
#if HAS_ASYNC
        __builtin_amdgcn_global_load_async_to_lds_b128((gvsi4*)(gA0 + k), (lvsi4*)&sA[buf][offA0], 0, 0);
        __builtin_amdgcn_global_load_async_to_lds_b128((gvsi4*)(gA1 + k), (lvsi4*)&sA[buf][offA1], 0, 0);
        __builtin_amdgcn_global_load_async_to_lds_b128((gvsi4*)(gB0 + k), (lvsi4*)&sB[buf][offA0], 0, 0);
        __builtin_amdgcn_global_load_async_to_lds_b128((gvsi4*)(gB1 + k), (lvsi4*)&sB[buf][offA1], 0, 0);
#else
        *(v8bf*)&sA[buf][offA0] = *(const v8bf*)(gA0 + k);
        *(v8bf*)&sA[buf][offA1] = *(const v8bf*)(gA1 + k);
        *(v8bf*)&sB[buf][offA0] = *(const v8bf*)(gB0 + k);
        *(v8bf*)&sB[buf][offA1] = *(const v8bf*)(gB1 + k);
#endif
    };

    fill(0, 0);   // prologue: K-step 0 into buffer 0

    for (int kb = 0; kb < NK; ++kb) {
        int cur = kb & 1;
#if HAS_ASYNC
        __builtin_amdgcn_s_wait_asynccnt(0);   // this wave's fill of `cur` done
#endif
        __syncthreads();                       // all waves' fills done; prev compute done
        if (kb + 1 < NK) fill(cur ^ 1, (kb + 1) * TK);

        // ---- B fragments for this wave's two 16-col slices ----
        v16bf bfrag[2];
#pragma unroll
        for (int n = 0; n < 2; ++n) {
            int colL = wn * 32 + n * 16 + r16;
            v8bf c0 = *(const v8bf*)&sB[cur][colL * LDT + 8 * half];
            v8bf c1 = *(const v8bf*)&sB[cur][colL * LDT + 16 + 8 * half];
            bfrag[n] = cat8(c0, c1);
        }
        // ---- A fragments + WMMA ----
#pragma unroll
        for (int m = 0; m < 4; ++m) {
            int rowL = wm * 64 + m * 16 + r16;
            v8bf a0 = *(const v8bf*)&sA[cur][rowL * LDT + 8 * half];
            v8bf a1 = *(const v8bf*)&sA[cur][rowL * LDT + 16 + 8 * half];
            v16bf afrag = cat8(a0, a1);
#pragma unroll
            for (int n = 0; n < 2; ++n) {
                acc[m][n] = __builtin_amdgcn_wmma_f32_16x16x32_bf16(
                    false, afrag, false, bfrag[n], (short)0, acc[m][n], false, false);
            }
        }
    }

    // ---- epilogue: out[token, f] += gate * (acc + bias_e[f]) ----
#pragma unroll
    for (int n = 0; n < 2; ++n) {
        int f = n0 + wn * 32 + n * 16 + r16;
        float bias = eb[(size_t)e * DD + f];
#pragma unroll
        for (int m = 0; m < 4; ++m) {
#pragma unroll
            for (int v = 0; v < 8; ++v) {
                int rowL = wm * 64 + m * 16 + v + 8 * half;   // C layout: M = v + 8*(lane/16)
                int rg   = m0 + rowL;
                if (rg < cnt) {
                    float g   = sgate[rowL];
                    int   tok = stok[rowL];
                    atomicAdd(&out[(size_t)tok * DD + f], g * (acc[m][n][v] + bias));
                }
            }
        }
    }
}

// ---------------------------------------------------------------------------
extern "C" void kernel_launch(void* const* d_in, const int* in_sizes, int n_in,
                              void* d_out, int out_size, void* d_ws, size_t ws_size,
                              hipStream_t stream) {
    (void)in_sizes; (void)n_in; (void)ws_size;

    const float* x  = (const float*)d_in[0];
    const float* rw = (const float*)d_in[1];
    const float* rb = (const float*)d_in[2];
    const float* ew = (const float*)d_in[3];
    const float* eb = (const float*)d_in[4];
    float* out = (float*)d_out;

    const size_t XB_BYTES = (size_t)BS * DD * sizeof(__bf16);          // 33.5 MB
    const size_t WT_BYTES = (size_t)EE * DD * DD * sizeof(__bf16);     // 67 MB
    char* ws = (char*)d_ws;
    __bf16* xb        = (__bf16*)ws;
    __bf16* wT        = (__bf16*)(ws + XB_BYTES);
    int*    counts    = (int*)  (ws + XB_BYTES + WT_BYTES);
    int*    tok_list  = (int*)  (ws + XB_BYTES + WT_BYTES + 256);
    float*  gate_list = (float*)(ws + XB_BYTES + WT_BYTES + 256 + (size_t)EE * BS * 4);

    (void)hipMemsetAsync(counts, 0, 256, stream);
    (void)hipMemsetAsync(d_out, 0, (size_t)out_size * sizeof(float), stream);

    moe_conv_x<<<(BS * (size_t)DD) / (256 * 4), 256, 0, stream>>>(x, xb);
    moe_conv_wT<<<dim3(DD / 32, DD / 32, EE), 256, 0, stream>>>(ew, wT);
    moe_router<<<BS / 8, 256, 0, stream>>>(x, rw, rb, counts, tok_list, gate_list);
    moe_gemm<<<dim3(DD / TN, BS / TM, EE), 256, 0, stream>>>(
        xb, wT, counts, tok_list, gate_list, eb, out);
}
